// Global_SlidingWindow_Attention_20512763805788
// MI455X (gfx1250) — matrix operand order repaired and checked by reading, not compiled, not run on hardware
//
#include <hip/hip_runtime.h>

// ---------------------------------------------------------------------------
// Global + sliding-window attention for MI455X (gfx1250, wave32, WMMA).
// All four 4096x1024x1024 GEMMs and both attention matmuls run through
// v_wmma_f32_16x16x32_f16 (f16 inputs, f32 accumulation).
// GEMM: 64x64 per-wave tile (16 WMMA / 16 b128 loads per K-step = 32 FLOP/B
// from L2). Fragment addressing uses two base pointers + 24-bit immediate
// offsets, and __launch_bounds__(256,1) raises the VGPR budget so the 128
// accumulator VGPRs live in registers (no scratch spills).
// ---------------------------------------------------------------------------

typedef _Float16 f16;
typedef __attribute__((ext_vector_type(16))) _Float16 v16h;
typedef __attribute__((ext_vector_type(8)))  _Float16 v8h;
typedef __attribute__((ext_vector_type(8)))  float    v8f;

#define HIDDEN 1024
#define HEADS  16
#define HDIM   64

union AFrag { v16h v; v8h h[2]; };

__device__ __forceinline__ v8f vzero() {
  v8f z = {0.f, 0.f, 0.f, 0.f, 0.f, 0.f, 0.f, 0.f};
  return z;
}

__device__ __forceinline__ v8f wmma_f16(v16h a, v16h b, v8f c) {
  // (neg_a, A, neg_b, B, c_mod, C, reuse_a, reuse_b)
  return __builtin_amdgcn_wmma_f32_16x16x32_f16(false, a, false, b, (short)0, c,
                                                false, false);
}

// --------------------------- f32 -> f16 convert ----------------------------
__global__ void k_cvt(const float* __restrict__ x, f16* __restrict__ y, int n) {
  int i = (blockIdx.x * blockDim.x + threadIdx.x) * 4;
  if (i + 3 < n) {
    float4 f = *(const float4*)(x + i);
    y[i + 0] = (f16)f.x;
    y[i + 1] = (f16)f.y;
    y[i + 2] = (f16)f.z;
    y[i + 3] = (f16)f.w;
  }
}

// ------------------- W [K,N] f32  ->  Wt [N,K] f16 (LDS tiled) -------------
__global__ void k_transpose(const float* __restrict__ w, f16* __restrict__ wt) {
  __shared__ float tile[32][33];
  int nb = blockIdx.x * 32, kb = blockIdx.y * 32;
  int tx = threadIdx.x, ty = threadIdx.y;  // block (32,8)
#pragma unroll
  for (int i = 0; i < 32; i += 8)
    tile[ty + i][tx] = w[(size_t)(kb + ty + i) * HIDDEN + (nb + tx)];
  __syncthreads();
#pragma unroll
  for (int i = 0; i < 32; i += 8)
    wt[(size_t)(nb + ty + i) * HIDDEN + (kb + tx)] = (f16)tile[tx][ty + i];
}

// ------------------------------- GEMM --------------------------------------
// C[M,N] = A[M,1024] * Wt[N,1024]^T + bias.  One wave: 64(M) x 64(N) tile.
// mode 0: out f16 [B,H,S,D]      (Q, K)
// mode 1: out f16 [B,H,D,S]      (V transposed for P*V B-fragments)
// mode 2: out f32 [M,N]          (final projection -> d_out)
__global__ void __launch_bounds__(256, 1)
k_gemm(const f16* __restrict__ A, const f16* __restrict__ Wt,
       const float* __restrict__ bias, void* __restrict__ outp,
       int sShift, int mode) {
  const int lane = threadIdx.x & 31;
  const int wave = threadIdx.x >> 5;
  const int lr = lane & 15, lh = lane >> 4;
  const int mbase = blockIdx.y * 64;
  const int nbase = blockIdx.x * 512 + wave * 64;
  const int S = 1 << sShift;

  v8f c[4][4];
#pragma unroll
  for (int mi = 0; mi < 4; ++mi)
#pragma unroll
    for (int ni = 0; ni < 4; ++ni) c[mi][ni] = vzero();

  // Single base pointer per stream; row tiles are +mi*32768B / +ni*32768B,
  // expressible as global_load immediate offsets.
  const f16* aptr = A + (size_t)(mbase + lr) * HIDDEN + lh * 16;
  const f16* wptr = Wt + (size_t)(nbase + lr) * HIDDEN + lh * 16;

  for (int kt = 0; kt < HIDDEN; kt += 32) {
    AFrag a[4];
#pragma unroll
    for (int mi = 0; mi < 4; ++mi) {
      a[mi].h[0] = *(const v8h*)(aptr + mi * 16 * HIDDEN);
      a[mi].h[1] = *(const v8h*)(aptr + mi * 16 * HIDDEN + 8);
    }
    v16h b[4];
#pragma unroll
    for (int ni = 0; ni < 4; ++ni)
      b[ni] = *(const v16h*)(wptr + ni * 16 * HIDDEN);
#pragma unroll
    for (int mi = 0; mi < 4; ++mi)
#pragma unroll
      for (int ni = 0; ni < 4; ++ni)
        c[mi][ni] = wmma_f16(a[mi].v, b[ni], c[mi][ni]);
    aptr += 32;
    wptr += 32;
  }

#pragma unroll
  for (int ni = 0; ni < 4; ++ni) {
    int n = nbase + ni * 16 + lr;
    float bb = bias[n];
    int hh = n >> 6, dd = n & 63;
#pragma unroll
    for (int mi = 0; mi < 4; ++mi) {
#pragma unroll
      for (int r = 0; r < 8; ++r) {
        int m = mbase + mi * 16 + r + 8 * lh;
        float val = c[mi][ni][r] + bb;
        int bi = m >> sShift, si = m & (S - 1);
        if (mode == 0) {
          ((f16*)outp)[((size_t)(bi * HEADS + hh) * S + si) * HDIM + dd] =
              (f16)val;
        } else if (mode == 1) {
          ((f16*)outp)[((size_t)(bi * HEADS + hh) * HDIM + dd) * S + si] =
              (f16)val;
        } else {
          ((float*)outp)[(size_t)m * HIDDEN + n] = val;
        }
      }
    }
  }
}

// --------------------------- attention -------------------------------------
// One wave per (b, h, 16-query tile); 4 waves per block.
// Flash-style online softmax over 32-key chunks; QK^T and P*V via WMMA.
__global__ void k_attn(const f16* __restrict__ Q, const f16* __restrict__ K,
                       const f16* __restrict__ Vt, f16* __restrict__ ctx,
                       const int* __restrict__ wptr, int S) {
  const int lane = threadIdx.x & 31;
  const int wave = threadIdx.x >> 5;
  const int lr = lane & 15, lh = lane >> 4;
  const int h = blockIdx.y, b = blockIdx.z;
  const int qbase = (blockIdx.x * 4 + wave) * 16;
  const int hw = wptr[0] >> 1;  // half window (=128)

  __shared__ f16 plds_all[4][16 * 32];
  f16* plds = plds_all[wave];

  const size_t bh = (size_t)b * HEADS + h;
  const f16* Qp = Q + (bh * S + qbase) * HDIM;
  const f16* Kp = K + bh * S * HDIM;
  const f16* Vp = Vt + bh * HDIM * S;

  // Q A-fragments for d-chunks [0,32) and [32,64)
  AFrag qa[2];
  {
    const f16* qr = Qp + lr * HDIM;
#pragma unroll
    for (int dc = 0; dc < 2; ++dc) {
      qa[dc].h[0] = *(const v8h*)(qr + dc * 32 + lh * 16);
      qa[dc].h[1] = *(const v8h*)(qr + dc * 32 + lh * 16 + 8);
    }
  }

  v8f acc[4];
#pragma unroll
  for (int dt = 0; dt < 4; ++dt) acc[dt] = vzero();
  float m[8], ell[8];
#pragma unroll
  for (int r = 0; r < 8; ++r) { m[r] = -3.0e38f; ell[r] = 0.f; }

  int start = qbase - hw;      if (start < 0) start = 0;
  int end   = qbase + 15 + hw; if (end > S - 1) end = S - 1;
  int c0 = start & ~31;
  int nch = ((end - c0) >> 5) + 1;
  int extra = (c0 > 0) ? 1 : 0;  // prepend chunk 0 for the global token

  for (int ci = 0; ci < nch + extra; ++ci) {
    int kc = (ci < extra) ? 0 : c0 + ((ci - extra) << 5);

    // ---- scores: 16 x 32 tile via 4 WMMAs ----
    v8f st[2];
    st[0] = vzero(); st[1] = vzero();
#pragma unroll
    for (int t = 0; t < 2; ++t) {
      const f16* kr = Kp + (size_t)(kc + t * 16 + lr) * HDIM;
#pragma unroll
      for (int dc = 0; dc < 2; ++dc) {
        v16h kb = *(const v16h*)(kr + dc * 32 + lh * 16);
        st[t] = wmma_f16(qa[dc].v, kb, st[t]);
      }
    }

    // ---- mask + scale + online softmax ----
#pragma unroll
    for (int r = 0; r < 8; ++r) {
      int i = qbase + r + 8 * lh;
      int j0 = kc + lr;
      int j1 = kc + 16 + lr;
      bool v0 = ((j0 - i <= hw) && (i - j0 <= hw)) || (j0 == 0);
      bool v1 = ((j1 - i <= hw) && (i - j1 <= hw)) || (j1 == 0);
      float s0 = v0 ? st[0][r] * 0.125f : -1e9f;  // 1/sqrt(64)
      float s1 = v1 ? st[1][r] * 0.125f : -1e9f;

      float rm = fmaxf(s0, s1);
#pragma unroll
      for (int sft = 1; sft < 16; sft <<= 1)
        rm = fmaxf(rm, __shfl_xor(rm, sft, 32));
      float newm = fmaxf(m[r], rm);
      float alpha = __expf(m[r] - newm);
      m[r] = newm;
      float p0 = __expf(s0 - newm);
      float p1 = __expf(s1 - newm);
      float rs = p0 + p1;
#pragma unroll
      for (int sft = 1; sft < 16; sft <<= 1) rs += __shfl_xor(rs, sft, 32);
      ell[r] = ell[r] * alpha + rs;
#pragma unroll
      for (int dt = 0; dt < 4; ++dt) acc[dt][r] *= alpha;
      plds[(r + 8 * lh) * 32 + lr] = (f16)p0;
      plds[(r + 8 * lh) * 32 + 16 + lr] = (f16)p1;
    }

    // ---- reshape P (C-layout -> A-fragment) through LDS ----
    AFrag pa;
    pa.h[0] = *(const v8h*)(plds + lr * 32 + lh * 16);
    pa.h[1] = *(const v8h*)(plds + lr * 32 + lh * 16 + 8);

    // ---- context += P @ V : 4 WMMAs ----
#pragma unroll
    for (int dt = 0; dt < 4; ++dt) {
      v16h vb = *(const v16h*)(Vp + (size_t)(dt * 16 + lr) * S + kc + lh * 16);
      acc[dt] = wmma_f16(pa.v, vb, acc[dt]);
    }
  }

  // ---- normalize and store ctx as f16 [M, HIDDEN] for the final GEMM ----
#pragma unroll
  for (int dt = 0; dt < 4; ++dt) {
#pragma unroll
    for (int r = 0; r < 8; ++r) {
      int q = qbase + r + 8 * lh;
      int d = dt * 16 + lr;
      float o = acc[dt][r] / ell[r];
      ctx[(size_t)(b * S + q) * HIDDEN + h * HDIM + d] = (f16)o;
    }
  }
}

// ---------------------------------------------------------------------------
extern "C" void kernel_launch(void* const* d_in, const int* in_sizes, int n_in,
                              void* d_out, int out_size, void* d_ws,
                              size_t ws_size, hipStream_t stream) {
  const float* X  = (const float*)d_in[0];
  const float* Wq = (const float*)d_in[1];
  const float* bq = (const float*)d_in[2];
  const float* Wk = (const float*)d_in[3];
  const float* bk = (const float*)d_in[4];
  const float* Wv = (const float*)d_in[5];
  const float* bv = (const float*)d_in[6];
  const float* Wo = (const float*)d_in[7];
  const float* bo = (const float*)d_in[8];
  const int*  win = (const int*)d_in[9];

  const int B = 2;
  const int S = in_sizes[0] / (B * HIDDEN);  // power of two (2048)
  const int M = B * S;
  int sShift = 0;
  while ((1 << sShift) < S) ++sShift;

  char* p = (char*)d_ws;
  f16* Xh  = (f16*)p; p += (size_t)M * HIDDEN * sizeof(f16);
  f16* Wqt = (f16*)p; p += (size_t)HIDDEN * HIDDEN * sizeof(f16);
  f16* Wkt = (f16*)p; p += (size_t)HIDDEN * HIDDEN * sizeof(f16);
  f16* Wvt = (f16*)p; p += (size_t)HIDDEN * HIDDEN * sizeof(f16);
  f16* Wot = (f16*)p; p += (size_t)HIDDEN * HIDDEN * sizeof(f16);
  f16* Qh  = (f16*)p; p += (size_t)M * HIDDEN * sizeof(f16);
  f16* Kh  = (f16*)p; p += (size_t)M * HIDDEN * sizeof(f16);
  f16* Vth = (f16*)p; p += (size_t)M * HIDDEN * sizeof(f16);
  f16* Ch  = (f16*)p; p += (size_t)M * HIDDEN * sizeof(f16);

  // 1) activations f32 -> f16
  k_cvt<<<(M * HIDDEN / 4 + 255) / 256, 256, 0, stream>>>(X, Xh, M * HIDDEN);

  // 2) weights f32 [K,N] -> f16 [N,K]
  dim3 tg(HIDDEN / 32, HIDDEN / 32), tb(32, 8);
  k_transpose<<<tg, tb, 0, stream>>>(Wq, Wqt);
  k_transpose<<<tg, tb, 0, stream>>>(Wk, Wkt);
  k_transpose<<<tg, tb, 0, stream>>>(Wv, Wvt);
  k_transpose<<<tg, tb, 0, stream>>>(Wo, Wot);

  // 3) QKV projections (WMMA GEMMs)
  dim3 gg(HIDDEN / 512, M / 64);
  k_gemm<<<gg, 256, 0, stream>>>(Xh, Wqt, bq, Qh, sShift, 0);
  k_gemm<<<gg, 256, 0, stream>>>(Xh, Wkt, bk, Kh, sShift, 0);
  k_gemm<<<gg, 256, 0, stream>>>(Xh, Wvt, bv, Vth, sShift, 1);

  // 4) sliding-window + global-token attention (WMMA flash)
  k_attn<<<dim3(S / 64, HEADS, B), 128, 0, stream>>>(Qh, Kh, Vth, Ch, win, S);

  // 5) output projection -> f32 d_out
  k_gemm<<<gg, 256, 0, stream>>>(Ch, Wot, bo, d_out, sShift, 2);
}
